// MinGRU_74268574483142
// MI455X (gfx1250) — matrix operand order repaired and checked by reading, not compiled, not run on hardware
//
#include <hip/hip_runtime.h>

typedef __bf16 bf16;
typedef __attribute__((ext_vector_type(16))) __bf16 v16bf;
typedef __attribute__((ext_vector_type(8)))  __bf16 v8bf;
typedef __attribute__((ext_vector_type(4)))  __bf16 v4bf;
typedef __attribute__((ext_vector_type(8)))  float   v8f;
typedef __attribute__((ext_vector_type(4)))  float   v4f;

#define TILE_M 128
#define TILE_N 128
#define TILE_K 32
#define LDS_STRIDE 40   // 32 + 8 bf16 pad; 80B row stride, 16B-aligned chunks

union FragAB { v16bf v; v8bf h[2]; };

// ---------------------------------------------------------------------------
// fp32 -> bf16 streaming conversion (v_cvt_pk_bf16_f32), 4 elems/thread
// ---------------------------------------------------------------------------
__global__ __launch_bounds__(256)
void cvt_f32_bf16_kernel(const float* __restrict__ in, bf16* __restrict__ out,
                         int n4)
{
    int i = blockIdx.x * blockDim.x + threadIdx.x;
    if (i >= n4) return;
    v4f v = *(const v4f*)(in + (size_t)i * 4);
    v4bf o;
    o[0] = (bf16)v[0]; o[1] = (bf16)v[1]; o[2] = (bf16)v[2]; o[3] = (bf16)v[3];
    *(v4bf*)(out + (size_t)i * 4) = o;
}

// ---------------------------------------------------------------------------
// Async global->LDS copy (CDNA5 TDM-adjacent path, ASYNCcnt-tracked).
// GVS mode: SADDR = 64-bit sgpr base, VADDR = 32-bit per-lane byte offset,
// VDST = per-lane LDS byte address (low 32 bits of flat LDS address).
// ---------------------------------------------------------------------------
__device__ __forceinline__
void async_copy_b128(const bf16* lds_dst, const bf16* gbase, unsigned gbyteoff)
{
    unsigned lds_addr = (unsigned)(uintptr_t)lds_dst;   // addr[31:0] = LDS offset
    unsigned long long base = (unsigned long long)gbase;
    asm volatile("global_load_async_to_lds_b128 %0, %1, %2"
                 :: "v"(lds_addr), "v"(gbyteoff), "s"(base)
                 : "memory");
}

// ---------------------------------------------------------------------------
// D = A(MxK,bf16) * B(NxK,bf16)^T, fp32 accumulate via v_wmma_f32_16x16x32_bf16.
// 128x128 block tile, 8 waves (2x4), 64x32 per wave = 8 accumulators.
// LDS double-buffered, filled by async global->LDS b128 (no VGPR staging).
// SPLIT=1: cols [0,halfN) -> bf16 hidden, [halfN,N) -> sigmoid -> bf16 gate.
// SPLIT=0: fp32 out [M,N].
// ---------------------------------------------------------------------------
template<bool SPLIT>
__global__ __launch_bounds__(256)
void wmma_gemm_async_kernel(const bf16* __restrict__ Ap,
                            const bf16* __restrict__ Bp,
                            bf16* __restrict__ outHidden,
                            bf16* __restrict__ outGate,
                            float* __restrict__ outF,
                            int M, int N, int K, int halfN)
{
    __shared__ bf16 As[2][TILE_M][LDS_STRIDE];
    __shared__ bf16 Bs[2][TILE_N][LDS_STRIDE];

    const int tid   = threadIdx.x;
    const int lane  = tid & 31;
    const int wave  = tid >> 5;
    const int wm    = wave >> 2;      // 0..1 -> 64 rows each
    const int wn    = wave & 3;       // 0..3 -> 32 cols each
    const int mBase = wm * 64;
    const int nBase = wn * 32;
    const int lhalf = lane >> 4;      // 0|1
    const int lrow  = lane & 15;

    const int rowA0 = blockIdx.y * TILE_M;
    const int rowB0 = blockIdx.x * TILE_N;

    v8f acc[4][2] = {};

    // Stage one 128x32 bf16 tile of A and B: 512 x 16B chunks each, 256
    // threads x 2 chunks; 4 async instructions per wave per stage.
    auto stage = [&](int buf, int k0) {
        #pragma unroll
        for (int i = 0; i < 2; ++i) {
            int c   = tid + i * 256;      // 0..511
            int row = c >> 2;             // 0..127
            int kc  = (c & 3) * 8;        // element offset: 0,8,16,24 (16B chunks)
            unsigned aoff = (unsigned)(((size_t)(rowA0 + row) * K + k0 + kc) * sizeof(bf16));
            async_copy_b128(&As[buf][row][kc], Ap, aoff);
            unsigned boff = (unsigned)(((size_t)(rowB0 + row) * K + k0 + kc) * sizeof(bf16));
            async_copy_b128(&Bs[buf][row][kc], Bp, boff);
        }
    };

    auto compute = [&](int buf) {
        FragAB a[4], b[2];
        // A 16x32 bf16 layout: lanes 0-15 hold K 0-7 & 16-23, lanes 16-31
        // hold K 8-15 & 24-31 (ISA 7.12.2)
        #pragma unroll
        for (int i = 0; i < 4; ++i) {
            const bf16* p = &As[buf][mBase + i * 16 + lrow][0];
            a[i].h[0] = *(const v8bf*)(p + lhalf * 8);
            a[i].h[1] = *(const v8bf*)(p + 16 + lhalf * 8);
        }
        // B 32x16 bf16 layout: lanes 0-15 hold K 0-7 & 16-23, lanes 16-31
        // hold K 8-15 & 24-31 (matches A)
        #pragma unroll
        for (int j = 0; j < 2; ++j) {
            const bf16* p = &Bs[buf][nBase + j * 16 + lrow][0];
            b[j].h[0] = *(const v8bf*)(p + lhalf * 8);
            b[j].h[1] = *(const v8bf*)(p + 16 + lhalf * 8);
        }
        #pragma unroll
        for (int i = 0; i < 4; ++i)
            #pragma unroll
            for (int j = 0; j < 2; ++j)
                acc[i][j] = __builtin_amdgcn_wmma_f32_16x16x32_bf16(
                    false, a[i].v, false, b[j].v, (short)0, acc[i][j],
                    false, false);
    };

    stage(0, 0);
    int buf = 0;
    for (int k0 = TILE_K; k0 < K; k0 += TILE_K) {
        stage(buf ^ 1, k0);                               // prefetch next slice
        // wait for the older stage (4 newest async ops may remain in flight)
        asm volatile("s_wait_asynccnt 4" ::: "memory");
        __syncthreads();                                  // buf resident for all waves
        compute(buf);
        __syncthreads();                                  // reads done before restage
        buf ^= 1;
    }
    asm volatile("s_wait_asynccnt 0" ::: "memory");
    __syncthreads();
    compute(buf);

    // Epilogue.  C/D 16x16 f32 layout: vgpr r -> M = r + 8*lhalf, N = lane&15.
    #pragma unroll
    for (int i = 0; i < 4; ++i) {
        int rowg0 = rowA0 + mBase + i * 16 + lhalf * 8;
        #pragma unroll
        for (int j = 0; j < 2; ++j) {
            int coln = rowB0 + nBase + j * 16 + lrow;
            #pragma unroll
            for (int r = 0; r < 8; ++r) {
                float v   = acc[i][j][r];
                int  rowg = rowg0 + r;
                if (SPLIT) {
                    if (coln < halfN) {
                        outHidden[(size_t)rowg * halfN + coln] = (bf16)v;
                    } else {
                        float s = 1.0f / (1.0f + __expf(-v));
                        outGate[(size_t)rowg * halfN + (coln - halfN)] = (bf16)s;
                    }
                } else {
                    outF[(size_t)rowg * N + coln] = v;
                }
            }
        }
    }
}

// ---------------------------------------------------------------------------
// MinGRU linear recurrence with per-timestep reset.  One thread per (b,h)
// channel, fp32 carry in-register; writes bf16 h (GEMM2 operand) and fp32
// h_n.  is_init index is uniform per block -> scalar loads.  Prefetch T+16
// to hide the H-strided latency of the 4096-step sequential loop.
// ---------------------------------------------------------------------------
__global__ __launch_bounds__(256)
void mingru_scan_kernel(const bf16* __restrict__ gateb,
                        const bf16* __restrict__ hiddenb,
                        const float* __restrict__ is_init,
                        const float* __restrict__ h0,
                        bf16* __restrict__ hb,
                        float* __restrict__ hn,
                        int B, int T, int H)
{
    int gid = blockIdx.x * blockDim.x + threadIdx.x;
    if (gid >= B * H) return;
    int b = gid / H;
    int c = gid - b * H;

    const float h00 = h0[(size_t)b * H + c];
    float hprev     = h00;
    size_t base     = (size_t)b * T * H + c;
    size_t ibase    = (size_t)b * T;

    for (int t = 0; t < T; ++t) {
        size_t idx = base + (size_t)t * H;
        if (t + 16 < T) {
            __builtin_prefetch(&gateb[idx + (size_t)16 * H], 0, 0);
            __builtin_prefetch(&hiddenb[idx + (size_t)16 * H], 0, 0);
        }
        float g   = (float)gateb[idx];
        float hid = (float)hiddenb[idx];
        float r   = is_init[ibase + t];
        float carry = (r > 0.0f) ? h00 : hprev;   // reset to h_0
        float h = carry + g * (hid - carry);      // (1-g)*carry + g*hid
        hb[idx] = (bf16)h;
        hprev = h;
    }
    hn[(size_t)b * H + c] = hprev;
}

extern "C" void kernel_launch(void* const* d_in, const int* in_sizes, int n_in,
                              void* d_out, int out_size, void* d_ws, size_t ws_size,
                              hipStream_t stream) {
    (void)in_sizes; (void)n_in; (void)out_size; (void)ws_size;

    const float* x       = (const float*)d_in[0];   // (8,4096,1024)
    const float* W_hg    = (const float*)d_in[1];   // (2048,1024)
    const float* W_out   = (const float*)d_in[2];   // (1024,1024)
    const float* is_init = (const float*)d_in[3];   // (8,4096,1)
    const float* h0      = (const float*)d_in[4];   // (8,1,1024)

    const int B = 8, T = 4096, Din = 1024, H = 1024, Dout = 1024;
    const int M = B * T;                            // 32768
    const size_t BTH  = (size_t)M * H;              // 33,554,432
    const size_t NWHG = (size_t)2 * H * Din;        // 2,097,152
    const size_t NWO  = (size_t)Dout * H;           // 1,048,576

    // workspace (bf16): xb | W_hg | W_out | hidden | gate | h   (~262 MB)
    bf16* xb      = (bf16*)d_ws;
    bf16* wgb     = xb + BTH;
    bf16* wob     = wgb + NWHG;
    bf16* hiddenb = wob + NWO;
    bf16* gateb   = hiddenb + BTH;
    bf16* hb      = gateb + BTH;

    float* out = (float*)d_out;                     // (8,4096,1024)
    float* hn  = out + BTH;                         // (8,1,1024) tail

    // one-time fp32 -> bf16 conversion (makes GEMM operand re-reads bf16 + L2)
    cvt_f32_bf16_kernel<<<(int)(BTH / 4 + 255) / 256, 256, 0, stream>>>(x, xb, (int)(BTH / 4));
    cvt_f32_bf16_kernel<<<(int)(NWHG / 4 + 255) / 256, 256, 0, stream>>>(W_hg, wgb, (int)(NWHG / 4));
    cvt_f32_bf16_kernel<<<(int)(NWO / 4 + 255) / 256, 256, 0, stream>>>(W_out, wob, (int)(NWO / 4));

    // GEMM1: hg = x @ W_hg^T (M x 2H), fused sigmoid split epilogue
    dim3 g1((2 * H) / TILE_N, M / TILE_M);
    wmma_gemm_async_kernel<true><<<g1, 256, 0, stream>>>(
        xb, wgb, hiddenb, gateb, nullptr, M, 2 * H, Din, H);

    // Sequential scan over T, 8192 parallel channels
    mingru_scan_kernel<<<(B * H) / 256, 256, 0, stream>>>(
        gateb, hiddenb, is_init, h0, hb, hn, B, T, H);

    // GEMM2: out = h @ W_out^T (M x Dout), fp32 output
    dim3 g2(Dout / TILE_N, M / TILE_M);
    wmma_gemm_async_kernel<false><<<g2, 256, 0, stream>>>(
        hb, wob, nullptr, nullptr, out, M, Dout, H, 0);
}
